// _GeomBiasAttnLayer_48996986913259
// MI455X (gfx1250) — compile-verified, matrix operand order checked by reading, not run on hardware
//
#include <hip/hip_runtime.h>

typedef __attribute__((ext_vector_type(16))) _Float16 v16h;
typedef __attribute__((ext_vector_type(8)))  float    v8f;
typedef __attribute__((ext_vector_type(4)))  int      v4i;

// ---------------------------------------------------------------------------
// Problem constants (reference: B=2, N=256, M=1024, D=256, H=8, DH=32)
// ---------------------------------------------------------------------------
#define BB 2
#define NN 256
#define MM 1024
#define DD 256
#define HH 8
#define DHH 32

// CDNA5 async global->LDS copy path (ASYNCcnt). Guarded so the build can
// never regress if the builtins are absent on this toolchain.
#if __has_builtin(__builtin_amdgcn_global_load_async_to_lds_b128) && \
    __has_builtin(__builtin_amdgcn_s_wait_asynccnt)
#define USE_ASYNC_LDS 1
#else
#define USE_ASYNC_LDS 0
#endif

#if USE_ASYNC_LDS
__device__ __forceinline__ void async_g2l_b128(const void* g, void* l) {
    __builtin_amdgcn_global_load_async_to_lds_b128(
        (__attribute__((address_space(1))) v4i*)(g),
        (__attribute__((address_space(3))) v4i*)(l), 0, 0);
}
#endif

// ---------------------------------------------------------------------------
// f32 -> f16 conversion
// ---------------------------------------------------------------------------
__global__ void cvt_f16_kernel(const float* __restrict__ in, _Float16* __restrict__ out, int n) {
    int i = blockIdx.x * blockDim.x + threadIdx.x;
    if (i < n) out[i] = (_Float16)in[i];
}

// ---------------------------------------------------------------------------
// Generic WMMA GEMM: C[Mr x Nc] = act(A[Mr x K] @ Bm[K x Nc] + bias)
// A,Bm f16 row-major; C f32 (+ optional f16 mirror). Block = 128 threads
// (4 waves); each wave owns one 16x16 tile; block covers 16(M) x 64(N).
// Tiles staged to LDS with GLOBAL_LOAD_ASYNC_TO_LDS_B128 when available.
// ---------------------------------------------------------------------------
__global__ void wmma_gemm_kernel(const _Float16* __restrict__ A, int lda,
                                 const _Float16* __restrict__ Bm, int ldb,
                                 const float* __restrict__ bias,
                                 float* __restrict__ C, _Float16* __restrict__ C16,
                                 int ldc, int Mrows, int K, int act) {
    __shared__ __align__(16) _Float16 Ash[16 * 32];
    __shared__ __align__(16) _Float16 Bsh[32 * 64];
    const int lane  = threadIdx.x & 31;
    const int wave  = threadIdx.x >> 5;
    const int mbase = blockIdx.y * 16;
    const int nblk  = blockIdx.x * 64;
    const int kb    = (lane >> 4) * 8;   // K sub-block select (lanes 16..31 -> +8)
    const int l15   = lane & 15;
    v8f acc = {};
    for (int kk = 0; kk < K; kk += 32) {
#if USE_ASYNC_LDS
        // A tile 16x32 halves = 64 x 16B granules
        if (threadIdx.x < 64) {
            int r = threadIdx.x >> 2, c = (threadIdx.x & 3) * 8;
            async_g2l_b128(&A[(size_t)(mbase + r) * lda + kk + c], &Ash[r * 32 + c]);
        }
        // B tile 32x64 halves = 256 x 16B granules (2 per thread)
        for (int t = threadIdx.x; t < 256; t += blockDim.x) {
            int r = t >> 3, c = (t & 7) * 8;
            async_g2l_b128(&Bm[(size_t)(kk + r) * ldb + nblk + c], &Bsh[r * 64 + c]);
        }
        if (kk + 32 < K) {
            __builtin_prefetch(&A[(size_t)(mbase + l15) * lda + kk + 32], 0, 1);
            __builtin_prefetch(&Bm[(size_t)(kk + 32 + lane) * ldb + nblk], 0, 1);
        }
        __builtin_amdgcn_s_wait_asynccnt(0);
        __syncthreads();
#else
        for (int i = threadIdx.x; i < 16 * 32; i += blockDim.x) {
            int r = i >> 5, c = i & 31;
            Ash[i] = A[(size_t)(mbase + r) * lda + kk + c];
        }
        for (int i = threadIdx.x; i < 32 * 64; i += blockDim.x) {
            int r = i >> 6, c = i & 63;
            Bsh[i] = Bm[(size_t)(kk + r) * ldb + nblk + c];
        }
        if (kk + 32 < K) {
            __builtin_prefetch(&A[(size_t)(mbase + l15) * lda + kk + 32], 0, 1);
            __builtin_prefetch(&Bm[(size_t)(kk + 32 + lane) * ldb + nblk], 0, 1);
        }
        __syncthreads();
#endif
        v16h af, bf;
#pragma unroll
        for (int i = 0; i < 8; ++i) {
            int k = (i < 4) ? (kb + 2 * i) : (16 + kb + 2 * (i - 4));
            af[2 * i]     = Ash[l15 * 32 + k];
            af[2 * i + 1] = Ash[l15 * 32 + k + 1];
            bf[2 * i]     = Bsh[k * 64 + wave * 16 + l15];
            bf[2 * i + 1] = Bsh[(k + 1) * 64 + wave * 16 + l15];
        }
        acc = __builtin_amdgcn_wmma_f32_16x16x32_f16(false, af, false, bf,
                                                     (short)0, acc, false, false);
        __syncthreads();
    }
    const int rbase = mbase + ((lane >> 4) << 3);
    const int col   = nblk + wave * 16 + l15;
#pragma unroll
    for (int r = 0; r < 8; ++r) {
        int row = rbase + r;
        float v = acc[r];
        if (bias) v += bias[col];
        if (act == 1) v = fmaxf(v, 0.0f);
        size_t idx = (size_t)row * ldc + col;
        C[idx] = v;
        if (C16) C16[idx] = (_Float16)v;
    }
}

// ---------------------------------------------------------------------------
// Geo precompute: Ac = p_c @ Wg1[0:9,:] + bg1 ; Am = p_m @ Wg1[9:18,:]
// ---------------------------------------------------------------------------
__global__ void geo_pre_kernel(const float* __restrict__ p_c, const float* __restrict__ p_m,
                               const float* __restrict__ Wg1, const float* __restrict__ bg1,
                               float* __restrict__ Ac, float* __restrict__ Am,
                               int nC, int nM) {
    int i = blockIdx.x * blockDim.x + threadIdx.x;
    int totalC = nC * DD;
    if (i < totalC) {
        int row = i / DD, d = i % DD;
        float a = bg1[d];
#pragma unroll
        for (int j = 0; j < 9; ++j) a += p_c[row * 9 + j] * Wg1[j * DD + d];
        Ac[i] = a;
    } else {
        int i2 = i - totalC;
        if (i2 >= nM * DD) return;
        int row = i2 / DD, d = i2 % DD;
        float a = 0.0f;
#pragma unroll
        for (int j = 0; j < 9; ++j) a += p_m[row * 9 + j] * Wg1[(9 + j) * DD + d];
        Am[i2] = a;
    }
}

// ---------------------------------------------------------------------------
// geo_bias(b,n,m) = bg2 + sum_d relu(Ac[b,n,d] + Am[b,m,d]) * Wg2[d]
// grid: (MM/256, B*N); block 256
// ---------------------------------------------------------------------------
__global__ void geo_bias_kernel(const float* __restrict__ Ac, const float* __restrict__ Am,
                                const float* __restrict__ Wg2, const float* __restrict__ bg2,
                                float* __restrict__ geo) {
    __shared__ float acs[DD];
    __shared__ float wg[DD];
    int bn = blockIdx.y;          // b*N + n
    int b  = bn / NN;
    int m  = blockIdx.x * 256 + threadIdx.x;
    acs[threadIdx.x] = Ac[(size_t)bn * DD + threadIdx.x];
    wg[threadIdx.x]  = Wg2[threadIdx.x];
    __syncthreads();
    const float* amrow = Am + ((size_t)b * MM + m) * DD;
    float acc = bg2[0];
    for (int d = 0; d < DD; ++d)
        acc += fmaxf(acs[d] + amrow[d], 0.0f) * wg[d];
    geo[(size_t)bn * MM + m] = acc;
}

// ---------------------------------------------------------------------------
// Attention logits: S[bh,n,m] = scale * (Q_h . K_h) (+ beta*geo) (masked)
// K (head dim) = 32 = one WMMA. One wave per 16x16 (n,m) tile; block=4 waves.
// ---------------------------------------------------------------------------
__global__ void attn_logits_kernel(const _Float16* __restrict__ Q, int ldq, int qoff,
                                   const _Float16* __restrict__ Kf, int ldk, int koff,
                                   const float* __restrict__ geo,
                                   const unsigned char* __restrict__ amask,
                                   const unsigned char* __restrict__ mmask,
                                   const float* __restrict__ betaPtr, float scale,
                                   float* __restrict__ S, int Nq, int Mk) {
    int bh = blockIdx.z;
    int b = bh / HH, h = bh % HH;
    int nbase = blockIdx.y * 16;
    int wave = threadIdx.x >> 5;
    int lane = threadIdx.x & 31;
    int mbase = blockIdx.x * 64 + wave * 16;
    const int kb  = (lane >> 4) * 8;
    const int l15 = lane & 15;
    const _Float16* qrow = Q + (size_t)(b * Nq + nbase + l15) * ldq + qoff + h * DHH;
    const _Float16* krow = Kf + (size_t)(b * Mk + mbase + l15) * ldk + koff + h * DHH;
    v16h af, bf;
#pragma unroll
    for (int i = 0; i < 8; ++i) {
        int k = (i < 4) ? (kb + 2 * i) : (16 + kb + 2 * (i - 4));
        af[2 * i]     = qrow[k];
        af[2 * i + 1] = qrow[k + 1];
        bf[2 * i]     = krow[k];
        bf[2 * i + 1] = krow[k + 1];
    }
    v8f acc = {};
    acc = __builtin_amdgcn_wmma_f32_16x16x32_f16(false, af, false, bf,
                                                 (short)0, acc, false, false);
    float beta = betaPtr ? betaPtr[0] : 0.0f;
    int rbase = nbase + ((lane >> 4) << 3);
    int col = mbase + l15;
#pragma unroll
    for (int r = 0; r < 8; ++r) {
        int row = rbase + r;
        float v = acc[r] * scale;
        size_t gidx = ((size_t)(b * Nq + row)) * Mk + col;
        if (geo) v += beta * geo[gidx];
        if (amask) {
            bool ok = amask[gidx] && mmask[(size_t)b * Mk + col];
            if (!ok) v = -1e9f;
        }
        S[((size_t)bh * Nq + row) * Mk + col] = v;
    }
}

// ---------------------------------------------------------------------------
// Row softmax over M per (b,n) for all H heads; P (f16) out; optional
// head-mean accumulated into meanOut (B,N,M). Block = 256 threads.
// ---------------------------------------------------------------------------
__global__ void softmax_kernel(const float* __restrict__ S, _Float16* __restrict__ P,
                               float* __restrict__ meanOut, int Nq, int Mk) {
    __shared__ float red[256];
    int bn = blockIdx.x;
    int b = bn / Nq, n = bn % Nq;
    int tid = threadIdx.x;
    int elems = Mk / 256;
    float macc[4];
    for (int e = 0; e < elems; ++e) macc[e] = 0.0f;
    for (int h = 0; h < HH; ++h) {
        const float* row = S + ((size_t)(b * HH + h) * Nq + n) * Mk;
        _Float16* prow   = P + ((size_t)(b * HH + h) * Nq + n) * Mk;
        float lmax = -3.4e38f;
        for (int e = 0; e < elems; ++e) lmax = fmaxf(lmax, row[tid + e * 256]);
        red[tid] = lmax; __syncthreads();
        for (int s2 = 128; s2 > 0; s2 >>= 1) {
            if (tid < s2) red[tid] = fmaxf(red[tid], red[tid + s2]);
            __syncthreads();
        }
        float rmax = red[0]; __syncthreads();
        float ev[4]; float lsum = 0.0f;
        for (int e = 0; e < elems; ++e) {
            ev[e] = __expf(row[tid + e * 256] - rmax);
            lsum += ev[e];
        }
        red[tid] = lsum; __syncthreads();
        for (int s2 = 128; s2 > 0; s2 >>= 1) {
            if (tid < s2) red[tid] += red[tid + s2];
            __syncthreads();
        }
        float inv = 1.0f / red[0]; __syncthreads();
        for (int e = 0; e < elems; ++e) {
            float p = ev[e] * inv;
            prow[tid + e * 256] = (_Float16)p;
            macc[e] += p;
        }
    }
    if (meanOut) {
        for (int e = 0; e < elems; ++e)
            meanOut[(size_t)bn * Mk + tid + e * 256] = macc[e] * (1.0f / HH);
    }
}

// ---------------------------------------------------------------------------
// P @ V per head: ctx[b,n, h*32+d] = sum_m P[bh,n,m] * V[b,m, voff+h*32+d]
// block = 64 threads (2 waves, 2x16 cols = 32 = DH); K loop over Mk by 32.
// ---------------------------------------------------------------------------
__global__ void attn_pv_kernel(const _Float16* __restrict__ P,
                               const _Float16* __restrict__ V, int ldv, int voff,
                               float* __restrict__ ctx, _Float16* __restrict__ ctx16,
                               int Nq, int Mk) {
    int bh = blockIdx.z;
    int b = bh / HH, h = bh % HH;
    int nbase = blockIdx.y * 16;
    int wave = threadIdx.x >> 5;
    int lane = threadIdx.x & 31;
    int dbase = wave * 16;
    const int kb  = (lane >> 4) * 8;
    const int l15 = lane & 15;
    const _Float16* prow = P + ((size_t)bh * Nq + nbase + l15) * Mk;
    const _Float16* vcol = V + (size_t)b * Mk * ldv + voff + h * DHH + dbase + l15;
    v8f acc = {};
    for (int kk = 0; kk < Mk; kk += 32) {
        v16h af, bf;
#pragma unroll
        for (int i = 0; i < 8; ++i) {
            int k = (i < 4) ? (kb + 2 * i) : (16 + kb + 2 * (i - 4));
            af[2 * i]     = prow[kk + k];
            af[2 * i + 1] = prow[kk + k + 1];
            bf[2 * i]     = vcol[(size_t)(kk + k) * ldv];
            bf[2 * i + 1] = vcol[(size_t)(kk + k + 1) * ldv];
        }
        acc = __builtin_amdgcn_wmma_f32_16x16x32_f16(false, af, false, bf,
                                                     (short)0, acc, false, false);
    }
    int rbase = nbase + ((lane >> 4) << 3);
    int col = h * DHH + dbase + l15;
#pragma unroll
    for (int r = 0; r < 8; ++r) {
        size_t idx = ((size_t)(b * Nq + rbase + r)) * DD + col;
        ctx[idx] = acc[r];
        if (ctx16) ctx16[idx] = (_Float16)acc[r];
    }
}

// ---------------------------------------------------------------------------
// GRU gate fusion (torch order r,z,n): q_upd = (1-z)*n + z*h
// ---------------------------------------------------------------------------
__global__ void gru_kernel(const float* __restrict__ gi, const float* __restrict__ gh,
                           const float* __restrict__ hst,
                           float* __restrict__ qupd, _Float16* __restrict__ qupd16,
                           int total) {
    int i = blockIdx.x * blockDim.x + threadIdx.x;
    if (i >= total) return;
    int row = i / DD, c = i % DD;
    size_t base = (size_t)row * 3 * DD;
    float ir = gi[base + c], iz = gi[base + DD + c], inn = gi[base + 2 * DD + c];
    float hr = gh[base + c], hz = gh[base + DD + c], hn  = gh[base + 2 * DD + c];
    float r = 1.0f / (1.0f + __expf(-(ir + hr)));
    float z = 1.0f / (1.0f + __expf(-(iz + hz)));
    float n = tanhf(inn + r * hn);
    float hv = hst[i];
    float v = (1.0f - z) * n + z * hv;
    qupd[i] = v;
    qupd16[i] = (_Float16)v;
}

// ---------------------------------------------------------------------------
// out = LayerNorm(X + Y) * g + be ; block = 256 (= D) threads, one row each.
// ---------------------------------------------------------------------------
__global__ void add_ln_kernel(const float* __restrict__ X, const float* __restrict__ Y,
                              const float* __restrict__ g, const float* __restrict__ be,
                              float* __restrict__ out, _Float16* __restrict__ out16) {
    __shared__ float red[DD];
    int row = blockIdx.x, c = threadIdx.x;
    size_t idx = (size_t)row * DD + c;
    float x = X[idx] + Y[idx];
    red[c] = x; __syncthreads();
    for (int s = 128; s > 0; s >>= 1) { if (c < s) red[c] += red[c + s]; __syncthreads(); }
    float mu = red[0] * (1.0f / DD); __syncthreads();
    float d = x - mu;
    red[c] = d * d; __syncthreads();
    for (int s = 128; s > 0; s >>= 1) { if (c < s) red[c] += red[c + s]; __syncthreads(); }
    float var = red[0] * (1.0f / DD);
    float y = d * rsqrtf(var + 1e-5f) * g[c] + be[c];
    out[idx] = y;
    if (out16) out16[idx] = (_Float16)y;
}

// ---------------------------------------------------------------------------
// Host orchestration
// ---------------------------------------------------------------------------
extern "C" void kernel_launch(void* const* d_in, const int* in_sizes, int n_in,
                              void* d_out, int out_size, void* d_ws, size_t ws_size,
                              hipStream_t stream) {
    (void)in_sizes; (void)n_in; (void)out_size; (void)ws_size;
    const float* q    = (const float*)d_in[0];
    const float* k    = (const float*)d_in[1];
    const float* p_c  = (const float*)d_in[2];
    const float* p_m  = (const float*)d_in[3];
    const unsigned char* mem_mask  = (const unsigned char*)d_in[4];
    const unsigned char* attn_mask = (const unsigned char*)d_in[5];
    const float* Wq = (const float*)d_in[6];   const float* bq = (const float*)d_in[7];
    const float* Wk = (const float*)d_in[8];   const float* bk = (const float*)d_in[9];
    const float* Wv = (const float*)d_in[10];  const float* bv = (const float*)d_in[11];
    const float* Wg1 = (const float*)d_in[12]; const float* bg1 = (const float*)d_in[13];
    const float* Wg2 = (const float*)d_in[14]; const float* bg2 = (const float*)d_in[15];
    const float* beta = (const float*)d_in[16];
    const float* W_ih = (const float*)d_in[17]; const float* b_ih = (const float*)d_in[18];
    const float* W_hh = (const float*)d_in[19]; const float* b_hh = (const float*)d_in[20];
    const float* in_w = (const float*)d_in[21]; const float* in_b = (const float*)d_in[22];
    const float* out_w = (const float*)d_in[23]; const float* out_b = (const float*)d_in[24];
    const float* g1 = (const float*)d_in[25];  const float* be1 = (const float*)d_in[26];
    const float* Wf1 = (const float*)d_in[27]; const float* bf1 = (const float*)d_in[28];
    const float* Wf2 = (const float*)d_in[29]; const float* bf2 = (const float*)d_in[30];
    const float* g3 = (const float*)d_in[31];  const float* be3 = (const float*)d_in[32];

    char* wsb = (char*)d_ws;
    size_t off = 0;
    auto alloc = [&](size_t bytes) -> void* {
        void* p = wsb + off;
        off += (bytes + 255) & ~(size_t)255;
        return p;
    };
    const size_t BN = (size_t)BB * NN;        // 512
    const size_t BM = (size_t)BB * MM;        // 2048

    // f16 weights
    _Float16* hWq  = (_Float16*)alloc(DD * DD * 2);
    _Float16* hWk  = (_Float16*)alloc(DD * DD * 2);
    _Float16* hWv  = (_Float16*)alloc(DD * DD * 2);
    _Float16* hWih = (_Float16*)alloc(DD * 3 * DD * 2);
    _Float16* hWhh = (_Float16*)alloc(DD * 3 * DD * 2);
    _Float16* hInw = (_Float16*)alloc(DD * 3 * DD * 2);
    _Float16* hOutw = (_Float16*)alloc(DD * DD * 2);
    _Float16* hWf1 = (_Float16*)alloc(DD * 4 * DD * 2);
    _Float16* hWf2 = (_Float16*)alloc(4 * DD * DD * 2);
    // f16 activations
    _Float16* hq  = (_Float16*)alloc(BN * DD * 2);
    _Float16* hk  = (_Float16*)alloc(BM * DD * 2);
    _Float16* hqh = (_Float16*)alloc(BN * DD * 2);
    _Float16* hkh = (_Float16*)alloc(BM * DD * 2);
    _Float16* hvh = (_Float16*)alloc(BM * DD * 2);
    _Float16* hctx = (_Float16*)alloc(BN * DD * 2);
    _Float16* hqupd = (_Float16*)alloc(BN * DD * 2);
    _Float16* hqkv = (_Float16*)alloc(BN * 3 * DD * 2);
    _Float16* hsctx = (_Float16*)alloc(BN * DD * 2);
    _Float16* hqsa = (_Float16*)alloc(BN * DD * 2);
    _Float16* hffn1 = (_Float16*)alloc(BN * 4 * DD * 2);
    _Float16* Pc = (_Float16*)alloc((size_t)BB * HH * NN * MM * 2);
    _Float16* Ps = (_Float16*)alloc((size_t)BB * HH * NN * NN * 2);
    // f32 intermediates
    float* qpF = (float*)alloc(BN * DD * 4);
    float* kpF = (float*)alloc(BM * DD * 4);
    float* vpF = (float*)alloc(BM * DD * 4);
    float* Ac  = (float*)alloc(BN * DD * 4);
    float* Am  = (float*)alloc(BM * DD * 4);
    float* geo = (float*)alloc((size_t)BB * NN * MM * 4);
    float* Sc  = (float*)alloc((size_t)BB * HH * NN * MM * 4);
    float* Ss  = (float*)alloc((size_t)BB * HH * NN * NN * 4);
    float* ctxF = (float*)alloc(BN * DD * 4);
    float* giF = (float*)alloc(BN * 3 * DD * 4);
    float* ghF = (float*)alloc(BN * 3 * DD * 4);
    float* qupdF = (float*)alloc(BN * DD * 4);
    float* qkvF = (float*)alloc(BN * 3 * DD * 4);
    float* sctxF = (float*)alloc(BN * DD * 4);
    float* saF = (float*)alloc(BN * DD * 4);
    float* qsaF = (float*)alloc(BN * DD * 4);
    float* ffn1F = (float*)alloc(BN * 4 * DD * 4);
    float* ffn2F = (float*)alloc(BN * DD * 4);

    float* outFinal = (float*)d_out;                       // (B,N,D)
    float* outAttn  = outFinal + BN * DD;                  // (B,N,M)

    auto cvt = [&](const float* src, _Float16* dst, int n) {
        cvt_f16_kernel<<<(n + 255) / 256, 256, 0, stream>>>(src, dst, n);
    };
    auto gemm = [&](const _Float16* A, int lda, const _Float16* Bm, int ldb,
                    const float* bias, float* C, _Float16* C16,
                    int Mr, int K, int Nc, int act) {
        dim3 g(Nc / 64, Mr / 16, 1);
        wmma_gemm_kernel<<<g, 128, 0, stream>>>(A, lda, Bm, ldb, bias, C, C16, Nc, Mr, K, act);
    };
    const float scale = 0.17677669529663687f;  // 1/sqrt(32)

    // --- precision staging ---
    cvt(Wq, hWq, DD * DD);       cvt(Wk, hWk, DD * DD);     cvt(Wv, hWv, DD * DD);
    cvt(W_ih, hWih, DD * 3 * DD); cvt(W_hh, hWhh, DD * 3 * DD);
    cvt(in_w, hInw, DD * 3 * DD); cvt(out_w, hOutw, DD * DD);
    cvt(Wf1, hWf1, DD * 4 * DD);  cvt(Wf2, hWf2, 4 * DD * DD);
    cvt(q, hq, (int)(BN * DD));   cvt(k, hk, (int)(BM * DD));

    // --- cross-attn projections (WMMA) ---
    gemm(hq, DD, hWq, DD, bq, qpF, hqh, (int)BN, DD, DD, 0);
    gemm(hk, DD, hWk, DD, bk, kpF, hkh, (int)BM, DD, DD, 0);
    gemm(hk, DD, hWv, DD, bv, vpF, hvh, (int)BM, DD, DD, 0);

    // --- factorized geo MLP ---
    {
        int total = (int)((BN + BM) * DD);
        geo_pre_kernel<<<(total + 255) / 256, 256, 0, stream>>>(
            p_c, p_m, Wg1, bg1, Ac, Am, (int)BN, (int)BM);
        geo_bias_kernel<<<dim3(MM / 256, (unsigned)BN), 256, 0, stream>>>(Ac, Am, Wg2, bg2, geo);
    }

    // --- cross attention: logits -> softmax (+ head-mean out) -> P@V ---
    attn_logits_kernel<<<dim3(MM / 64, NN / 16, BB * HH), 128, 0, stream>>>(
        hqh, DD, 0, hkh, DD, 0, geo, attn_mask, mem_mask, beta, scale, Sc, NN, MM);
    softmax_kernel<<<(unsigned)BN, 256, 0, stream>>>(Sc, Pc, outAttn, NN, MM);
    attn_pv_kernel<<<dim3(1, NN / 16, BB * HH), 64, 0, stream>>>(
        Pc, hvh, DD, 0, ctxF, hctx, NN, MM);

    // --- GRU cell: x = ctx, h = q ---
    gemm(hctx, DD, hWih, 3 * DD, b_ih, giF, nullptr, (int)BN, DD, 3 * DD, 0);
    gemm(hq,   DD, hWhh, 3 * DD, b_hh, ghF, nullptr, (int)BN, DD, 3 * DD, 0);
    gru_kernel<<<(unsigned)((BN * DD + 255) / 256), 256, 0, stream>>>(
        giF, ghF, q, qupdF, hqupd, (int)(BN * DD));

    // --- self attention ---
    gemm(hqupd, DD, hInw, 3 * DD, in_b, qkvF, hqkv, (int)BN, DD, 3 * DD, 0);
    attn_logits_kernel<<<dim3(NN / 64, NN / 16, BB * HH), 128, 0, stream>>>(
        hqkv, 3 * DD, 0, hqkv, 3 * DD, DD, nullptr, nullptr, nullptr, nullptr,
        scale, Ss, NN, NN);
    softmax_kernel<<<(unsigned)BN, 256, 0, stream>>>(Ss, Ps, nullptr, NN, NN);
    attn_pv_kernel<<<dim3(1, NN / 16, BB * HH), 64, 0, stream>>>(
        Ps, hqkv, 3 * DD, 2 * DD, sctxF, hsctx, NN, NN);
    gemm(hsctx, DD, hOutw, DD, out_b, saF, nullptr, (int)BN, DD, DD, 0);
    add_ln_kernel<<<(unsigned)BN, DD, 0, stream>>>(qupdF, saF, g1, be1, qsaF, hqsa);

    // --- FFN + final LN -> output ---
    gemm(hqsa, DD, hWf1, 4 * DD, bf1, ffn1F, hffn1, (int)BN, DD, 4 * DD, 1);
    gemm(hffn1, 4 * DD, hWf2, DD, bf2, ffn2F, nullptr, (int)BN, 4 * DD, DD, 0);
    add_ln_kernel<<<(unsigned)BN, DD, 0, stream>>>(qsaF, ffn2F, g3, be3, outFinal, nullptr);
}
